// HierarchicalDenseCRF_32220844655463
// MI455X (gfx1250) — matrix operand order repaired and checked by reading, not compiled, not run on hardware
//
#include <hip/hip_runtime.h>
#include <math.h>

typedef __attribute__((ext_vector_type(16))) __bf16 v16bf;
typedef __attribute__((ext_vector_type(8)))  float  v8f;
typedef unsigned int v4u __attribute__((ext_vector_type(4)));
typedef int          v4i __attribute__((ext_vector_type(4)));
typedef int          v8i __attribute__((ext_vector_type(8)));

#define NIMG 8
#define NCH  8           // CRF classes / Q channels
#define GCH  6           // guidance channels (3 image + 3 fused)
#define ICH  14          // NCH + GCH conv input channels
#define KTAPS 9          // 3x3
#define KRED (ICH*KTAPS) // 126
#define TS   16
#define HALO (TS+2)

static __device__ __forceinline__ int imin(int a, int b) { return a < b ? a : b; }

// ---------------- align_corners=True bilinear helpers ----------------
static __device__ __forceinline__ void ac_axis(int o, int O, int I,
                                               int& i0, int& i1, float& f) {
  float pos = (O == 1) ? 0.f : ((float)o * (float)(I - 1) / (float)(O - 1));
  i0 = (int)floorf(pos);
  i1 = imin(i0 + 1, I - 1);
  f  = pos - (float)i0;
}

template <typename T>
static __device__ __forceinline__ float bisample(const T* plane, int hi, int wi,
                                                 int ho, int wo, int oy, int ox) {
  int y0, y1, x0, x1; float fy, fx;
  ac_axis(oy, ho, hi, y0, y1, fy);
  ac_axis(ox, wo, wi, x0, x1, fx);
  float v00 = (float)plane[(size_t)y0 * wi + x0];
  float v01 = (float)plane[(size_t)y0 * wi + x1];
  float v10 = (float)plane[(size_t)y1 * wi + x0];
  float v11 = (float)plane[(size_t)y1 * wi + x1];
  float r0 = v00 + (v01 - v00) * fx;
  float r1 = v10 + (v11 - v10) * fx;
  return r0 + (r1 - r0) * fy;
}

// ---------------- weight folding + A-fragment packing ----------------
// W_final[o][c'][t] = sum_c compat[o][c] * (bilateral[c][c'][t] + (c==c')*spatial[c][t])
// packed into V_WMMA_*_BF16 16x32 A-fragments in contiguous k order (k = 16*half + e), 4 k-steps.
__global__ void prep_weights_kernel(const float* __restrict__ spatial_w,   // (8,1,3,3)
                                    const float* __restrict__ bilateral_w, // (8,14,3,3)
                                    const float* __restrict__ compat_w,    // (8,8,1,1)
                                    __bf16* __restrict__ Afrag) {          // [4][32][16]
  int s = threadIdx.x >> 5;       // k-step 0..3
  int L = threadIdx.x & 31;       // lane
  int Mr = L & 15;                // output row
  int g  = L >> 4;                // lane half
  for (int e = 0; e < 16; ++e) {
    int v = e >> 1, par = e & 1;
    int kl = 16 * g + 2 * v + par;
    int k  = s * 32 + kl;
    float val = 0.f;
    if (Mr < NCH && k < KRED) {
      int cp = k / KTAPS, t = k % KTAPS;
      for (int c = 0; c < NCH; ++c) {
        float wv = bilateral_w[(c * ICH + cp) * KTAPS + t];
        if (cp == c) wv += spatial_w[c * KTAPS + t];
        val += compat_w[Mr * NCH + c] * wv;
      }
    }
    Afrag[(s * 32 + L) * 16 + e] = (__bf16)val;
  }
}

// ---------------- elementwise / memory-bound kernels ----------------
__global__ void zero_f32_kernel(float* __restrict__ p, int n) {
  int i = blockIdx.x * blockDim.x + threadIdx.x;
  if (i < n) p[i] = 0.f;
}

// 1x1 conv projection: (N,C,HW) x (3,C) + b -> (N,3,HW)
__global__ void project_kernel(const float* __restrict__ in,
                               const float* __restrict__ w,
                               const float* __restrict__ b,
                               float* __restrict__ out, int N, int C, int HW) {
  int idx = blockIdx.x * blockDim.x + threadIdx.x;
  if (idx >= N * HW) return;
  int n = idx / HW, p = idx % HW;
  const float* base = in + (size_t)n * C * HW + p;
  float a0 = b[0], a1 = b[1], a2 = b[2];
  for (int c = 0; c < C; ++c) {
    float x = base[(size_t)c * HW];
    a0 += w[0 * C + c] * x;
    a1 += w[1 * C + c] * x;
    a2 += w[2 * C + c] * x;
  }
  float* ob = out + (size_t)n * 3 * HW + p;
  ob[0]              = a0;
  ob[(size_t)HW]     = a1;
  ob[(size_t)2 * HW] = a2;
}

// bilinear-upsample proj (N,3,h,w) to (N,3,H,W) and accumulate
__global__ void upsample_add_kernel(const float* __restrict__ proj,
                                    float* __restrict__ fused,
                                    int N, int h, int w, int H, int W) {
  int idx = blockIdx.x * blockDim.x + threadIdx.x;
  int total = N * 3 * H * W;
  if (idx >= total) return;
  int x = idx % W, y = (idx / W) % H, c = (idx / (W * H)) % 3, n = idx / (W * H * 3);
  const float* plane = proj + ((size_t)n * 3 + c) * h * w;
  fused[idx] += bisample(plane, h, w, H, W, y, x);
}

// guidance = concat-part of X: write image & fused/4 into channels 8..13 of BOTH
// ping-pong X buffers (they persist across CRF iterations).
__global__ void guidance_kernel(const float* __restrict__ image,
                                const float* __restrict__ fused,
                                __bf16* __restrict__ Xa, __bf16* __restrict__ Xb,
                                int N, int HW) {
  int idx = blockIdx.x * blockDim.x + threadIdx.x;
  int total = N * GCH * HW;
  if (idx >= total) return;
  int p = idx % HW, c = (idx / HW) % GCH, n = idx / (HW * GCH);
  float v = (c < 3) ? image[((size_t)n * 3 + c) * HW + p]
                    : 0.25f * fused[((size_t)n * 3 + (c - 3)) * HW + p];
  size_t o = ((size_t)n * ICH + NCH + c) * HW + p;
  Xa[o] = (__bf16)v;
  Xb[o] = (__bf16)v;
}

// downsample guidance channels of full-res X into both half-res X buffers
__global__ void downsample_guid_kernel(const __bf16* __restrict__ Xfull,
                                       __bf16* __restrict__ Xda, __bf16* __restrict__ Xdb,
                                       int N, int H, int W, int hd, int wd) {
  int idx = blockIdx.x * blockDim.x + threadIdx.x;
  int total = N * GCH * hd * wd;
  if (idx >= total) return;
  int x = idx % wd, y = (idx / wd) % hd, c = (idx / (wd * hd)) % GCH, n = idx / (wd * hd * GCH);
  const __bf16* plane = Xfull + ((size_t)n * ICH + NCH + c) * H * W;
  __bf16 v = (__bf16)bisample(plane, H, W, hd, wd, y, x);
  size_t o = ((size_t)n * ICH + NCH + c) * hd * wd + (size_t)y * wd + x;
  Xda[o] = v;
  Xdb[o] = v;
}

// Q0 = softmax(unary) written into channels 0..7 of X
__global__ void softmax_init_kernel(const float* __restrict__ unary,
                                    __bf16* __restrict__ X, int N, int HW) {
  int idx = blockIdx.x * blockDim.x + threadIdx.x;
  if (idx >= N * HW) return;
  int n = idx / HW, p = idx % HW;
  const float* ub = unary + (size_t)n * NCH * HW + p;
  float v[NCH], m = -1e30f;
  #pragma unroll
  for (int c = 0; c < NCH; ++c) { v[c] = ub[(size_t)c * HW]; m = fmaxf(m, v[c]); }
  float sum = 0.f;
  #pragma unroll
  for (int c = 0; c < NCH; ++c) { v[c] = __expf(v[c] - m); sum += v[c]; }
  float inv = 1.f / sum;
  __bf16* ob = X + (size_t)n * ICH * HW + p;
  #pragma unroll
  for (int c = 0; c < NCH; ++c) ob[(size_t)c * HW] = (__bf16)(v[c] * inv);
}

__global__ void resize_f32_kernel(const float* __restrict__ in, float* __restrict__ out,
                                  int N, int C, int hi, int wi, int ho, int wo) {
  int idx = blockIdx.x * blockDim.x + threadIdx.x;
  int total = N * C * ho * wo;
  if (idx >= total) return;
  int x = idx % wo, y = (idx / wo) % ho, c = (idx / (wo * ho)) % C, n = idx / (wo * ho * C);
  const float* plane = in + ((size_t)n * C + c) * hi * wi;
  out[idx] = bisample(plane, hi, wi, ho, wo, y, x);
}

// out = 0.5 * (Qfull + upsample(Qdown)); Q lives in channels 0..7 of X buffers
__global__ void combine_kernel(const __bf16* __restrict__ Xfull,
                               const __bf16* __restrict__ Xdown,
                               float* __restrict__ out,
                               int N, int H, int W, int hd, int wd) {
  int idx = blockIdx.x * blockDim.x + threadIdx.x;
  int total = N * NCH * H * W;
  if (idx >= total) return;
  int x = idx % W, y = (idx / W) % H, c = (idx / (W * H)) % NCH, n = idx / (W * H * NCH);
  float qf = (float)Xfull[((size_t)n * ICH + c) * H * W + (size_t)y * W + x];
  const __bf16* plane = Xdown + ((size_t)n * ICH + c) * hd * wd;
  float qd = bisample(plane, hd, wd, H, W, y, x);
  out[idx] = 0.5f * (qf + qd);
}

// ---------------- the WMMA CRF iteration kernel ----------------
// One block = one 16x16 pixel tile. Interior tiles fetch the 18x18x14 bf16 halo
// tile with one Tensor-Data-Mover descriptor (tensor_load_to_lds + s_wait_tensorcnt);
// border tiles use a predicated manual load (zero padding). im2col in LDS, then
// 4x V_WMMA_F32_16X16X32_BF16 per 16-pixel column, lane-local softmax epilogue.
__global__ __launch_bounds__(128) void crf_iter_kernel(
    const __bf16* __restrict__ Xin,    // (N,14,H,W): Q(0..7) + guidance(8..13)
    const float*  __restrict__ unary,  // (N,8,H,W)
    const __bf16* __restrict__ Afrag,  // [4][32][16] packed A-fragments
    __bf16* __restrict__ Xout,         // Q channels rewritten; guidance pre-filled
    int H, int W) {
  __shared__ __attribute__((aligned(32))) __bf16 raw[ICH][HALO][HALO];  // 9 KB
  __shared__ __attribute__((aligned(32))) __bf16 col[TS * TS][64];      // 32 KB slab

  const int tid  = threadIdx.x;
  const int lane = tid & 31;
  const int wv   = tid >> 5;
  const int tx0  = blockIdx.x * TS;
  const int ty0  = blockIdx.y * TS;
  const int n    = blockIdx.z;
  const size_t HW = (size_t)H * W;
  const __bf16* xbase = Xin + (size_t)n * ICH * HW;

  // Preload the 4 k-step A fragments (weights), 32 B per lane per step.
  v16bf afr[4];
  #pragma unroll
  for (int s = 0; s < 4; ++s)
    afr[s] = *(const v16bf*)(Afrag + (size_t)(s * 32 + lane) * 16);

  const bool interior = (tx0 >= 1) && (ty0 >= 1) &&
                        (tx0 + TS + 1 <= W) && (ty0 + TS + 1 <= H);

#if __has_builtin(__builtin_amdgcn_tensor_load_to_lds)
  if (interior) {
    if (wv == 0) {
      // ---- Tensor DMA descriptor (CDNA5 ISA) ----
      unsigned long long ga =
          (unsigned long long)(const void*)(xbase + (size_t)(ty0 - 1) * W + (tx0 - 1));
      unsigned ldsa = (unsigned)(size_t)(void*)&raw[0][0][0];
      v4u g0 = { 1u,                                    // count=1, user mode
                 ldsa,                                  // lds_addr
                 (unsigned)(ga & 0xFFFFFFFFu),          // global_addr[31:0]
                 (unsigned)((ga >> 32) & 0x1FFFFFFu) | (2u << 30) }; // [56:32] | type=2
      const int TD = 1 << 30;  // huge tensor dims: interior tile -> no OOB clipping
      v8i g1;
      g1[0] = 0x10000;                         // data_size = 2 bytes
      g1[1] = (int)((TD & 0xFFFF) << 16);      // tensor_dim0[15:0]
      g1[2] = (int)((TD >> 16) & 0xFFFF) |     // tensor_dim0[31:16]
              (int)((TD & 0xFFFF) << 16);      // tensor_dim1[15:0]
      g1[3] = (int)((TD >> 16) & 0xFFFF) |     // tensor_dim1[31:16]
              (HALO << 16);                    // tile_dim0 = 18
      g1[4] = HALO | (ICH << 16);              // tile_dim1 = 18, tile_dim2 = 14
      g1[5] = W;                               // tensor_dim0_stride[31:0] (elements)
      g1[6] = (int)((HW & 0xFFFFu) << 16);     // stride0[47:32]=0 | stride1[15:0]
      g1[7] = (int)(HW >> 16);                 // tensor_dim1_stride[47:16] (= H*W)
      v4i g2 = { TD, 0, 0, 0 };                // tensor_dim2 huge; no iterate
      v4i g3 = { 0, 0, 0, 0 };
#if defined(__clang_major__) && __clang_major__ >= 23
      v8i gpad = { 0, 0, 0, 0, 0, 0, 0, 0 };
      __builtin_amdgcn_tensor_load_to_lds(g0, g1, g2, g3, gpad, 0);
#else
      __builtin_amdgcn_tensor_load_to_lds(g0, g1, g2, g3, 0);
#endif
      __builtin_amdgcn_s_wait_tensorcnt(0);
    }
    __syncthreads();
  } else
#endif
  {
    // border fallback: predicated loads with zero padding
    for (int i = tid; i < ICH * HALO * HALO; i += 128) {
      int c = i / (HALO * HALO);
      int r = i % (HALO * HALO);
      int yy = r / HALO, xx = r % HALO;
      int gy = ty0 + yy - 1, gx = tx0 + xx - 1;
      __bf16 v = (__bf16)0.0f;
      if (gy >= 0 && gy < H && gx >= 0 && gx < W)
        v = xbase[(size_t)c * HW + (size_t)gy * W + gx];
      raw[c][yy][xx] = v;
    }
    __syncthreads();
  }

  v8f acc[4] = {};

  // Two 64-wide K slabs (LDS stays at 41 KB -> multiple blocks per WGP).
  #pragma unroll
  for (int h = 0; h < 2; ++h) {
    // im2col slab build: incremental (c,ty,tx) tracking, no integer division.
    {
      int c  = (h == 0) ? 0 : 7;     // k = 64h -> c = k/9
      int ty = 0;
      int tx = (h == 0) ? 0 : 1;     // k = 64h -> t = k%9
      const int py0 = tid >> 4,          px0 = tid & 15;
      const int py1 = (tid + 128) >> 4,  px1 = (tid + 128) & 15;
      for (int kk = 0; kk < 64; ++kk) {
        __bf16 v0, v1;
        if (c < ICH) {
          v0 = raw[c][py0 + ty][px0 + tx];
          v1 = raw[c][py1 + ty][px1 + tx];
        } else {
          v0 = (__bf16)0.0f; v1 = v0;
        }
        col[tid][kk]       = v0;
        col[tid + 128][kk] = v1;
        if (++tx == 3) { tx = 0; if (++ty == 3) { ty = 0; ++c; } }
      }
    }
    __syncthreads();

    // WMMA chain. Wave wv owns tile rows 4*wv .. 4*wv+3.
    #pragma unroll
    for (int tt = 0; tt < 4; ++tt) {
      int trow = wv * 4 + tt;
      const __bf16* cb = &col[trow * 16 + (lane & 15)][(lane >> 4) * 16];
      #pragma unroll
      for (int sl = 0; sl < 2; ++sl) {
        v16bf bfr = *(const v16bf*)(cb + sl * 32);  // two ds_load_b128
        acc[tt] = __builtin_amdgcn_wmma_f32_16x16x32_bf16(
            false, afr[h * 2 + sl], false, bfr, (short)0, acc[tt], false, false);
      }
    }
    __syncthreads();
  }

  // Epilogue: lanes 0-15 hold message[M=0..7] for pixel (trow, lane):
  // lane-local softmax of (unary - message), store bf16 Q into Xout.
  if (lane < 16) {
    #pragma unroll
    for (int tt = 0; tt < 4; ++tt) {
      int trow = wv * 4 + tt;
      int gy = ty0 + trow, gx = tx0 + lane;
      const float* ub = unary + (size_t)n * NCH * HW + (size_t)gy * W + gx;
      float v[NCH], m = -1e30f;
      #pragma unroll
      for (int c = 0; c < NCH; ++c) {
        v[c] = ub[(size_t)c * HW] - acc[tt][c];
        m = fmaxf(m, v[c]);
      }
      float sum = 0.f;
      #pragma unroll
      for (int c = 0; c < NCH; ++c) { v[c] = __expf(v[c] - m); sum += v[c]; }
      float inv = 1.f / sum;
      __bf16* ob = Xout + (size_t)n * ICH * HW + (size_t)gy * W + gx;
      #pragma unroll
      for (int c = 0; c < NCH; ++c) ob[(size_t)c * HW] = (__bf16)(v[c] * inv);
    }
  }
}

// ---------------- host orchestration ----------------
extern "C" void kernel_launch(void* const* d_in, const int* in_sizes, int n_in,
                              void* d_out, int out_size, void* d_ws, size_t ws_size,
                              hipStream_t stream) {
  (void)in_sizes; (void)n_in; (void)out_size; (void)ws_size;
  const float* logits = (const float*)d_in[0];
  const float* image  = (const float*)d_in[1];
  const float* skip[4] = {(const float*)d_in[2], (const float*)d_in[3],
                          (const float*)d_in[4], (const float*)d_in[5]};
  const float* spatial_w   = (const float*)d_in[6];
  const float* bilateral_w = (const float*)d_in[7];
  const float* compat_w    = (const float*)d_in[8];
  const float* sw[4] = {(const float*)d_in[9],  (const float*)d_in[11],
                        (const float*)d_in[13], (const float*)d_in[15]};
  const float* sb[4] = {(const float*)d_in[10], (const float*)d_in[12],
                        (const float*)d_in[14], (const float*)d_in[16]};

  const int N = NIMG, H = 512, W = 512, Hd = 256, Wd = 256;
  const int HW = H * W, HWd = Hd * Wd;
  const int sh[4] = {256, 128, 64, 32};
  const int sc[4] = {64, 128, 256, 512};

  char* ws = (char*)d_ws;
  size_t off = 0;
  auto alloc = [&](size_t bytes) {
    size_t o = off;
    off = (off + bytes + 255) & ~(size_t)255;
    return o;
  };
  __bf16* Afrag   = (__bf16*)(ws + alloc(4 * 32 * 16 * sizeof(__bf16)));
  float*  fused   = (float*) (ws + alloc((size_t)N * 3 * HW * sizeof(float)));
  float*  logitsD = (float*) (ws + alloc((size_t)N * NCH * HWd * sizeof(float)));
  __bf16* Xa      = (__bf16*)(ws + alloc((size_t)N * ICH * HW  * sizeof(__bf16)));
  __bf16* Xb      = (__bf16*)(ws + alloc((size_t)N * ICH * HW  * sizeof(__bf16)));
  __bf16* XaD     = (__bf16*)(ws + alloc((size_t)N * ICH * HWd * sizeof(__bf16)));
  __bf16* XbD     = (__bf16*)(ws + alloc((size_t)N * ICH * HWd * sizeof(__bf16)));
  float*  proj    = (float*) (ws + alloc((size_t)N * 3 * 256 * 256 * sizeof(float)));

  const int TPB = 256;
  auto blocks = [&](int total) { return (total + TPB - 1) / TPB; };

  // 1) fold spatial + bilateral + compat weights into WMMA A-fragments
  prep_weights_kernel<<<1, 128, 0, stream>>>(spatial_w, bilateral_w, compat_w, Afrag);

  // 2) fused skip projection pyramid -> guidance channels of Xa/Xb
  zero_f32_kernel<<<blocks(N * 3 * HW), TPB, 0, stream>>>(fused, N * 3 * HW);
  for (int l = 0; l < 4; ++l) {
    int hw = sh[l] * sh[l];
    project_kernel<<<blocks(N * hw), TPB, 0, stream>>>(skip[l], sw[l], sb[l], proj,
                                                       N, sc[l], hw);
    upsample_add_kernel<<<blocks(N * 3 * HW), TPB, 0, stream>>>(proj, fused, N,
                                                                sh[l], sh[l], H, W);
  }
  guidance_kernel<<<blocks(N * GCH * HW), TPB, 0, stream>>>(image, fused, Xa, Xb, N, HW);

  // 3) half-res inputs
  downsample_guid_kernel<<<blocks(N * GCH * HWd), TPB, 0, stream>>>(Xa, XaD, XbD,
                                                                    N, H, W, Hd, Wd);
  resize_f32_kernel<<<blocks(N * NCH * HWd), TPB, 0, stream>>>(logits, logitsD, N, NCH,
                                                               H, W, Hd, Wd);

  // 4) full-res mean field: Q0 = softmax(logits); 2 WMMA+TDM iterations
  softmax_init_kernel<<<blocks(N * HW), TPB, 0, stream>>>(logits, Xa, N, HW);
  {
    dim3 g(W / TS, H / TS, N);
    crf_iter_kernel<<<g, 128, 0, stream>>>(Xa, logits, Afrag, Xb, H, W);
    crf_iter_kernel<<<g, 128, 0, stream>>>(Xb, logits, Afrag, Xa, H, W);
  }

  // 5) half-res mean field
  softmax_init_kernel<<<blocks(N * HWd), TPB, 0, stream>>>(logitsD, XaD, N, HWd);
  {
    dim3 g(Wd / TS, Hd / TS, N);
    crf_iter_kernel<<<g, 128, 0, stream>>>(XaD, logitsD, Afrag, XbD, Hd, Wd);
    crf_iter_kernel<<<g, 128, 0, stream>>>(XbD, logitsD, Afrag, XaD, Hd, Wd);
  }

  // 6) combine: out = (Q_full + up(Q_down)) / 2
  combine_kernel<<<blocks(N * NCH * HW), TPB, 0, stream>>>(Xa, XaD, (float*)d_out,
                                                           N, H, W, Hd, Wd);
}
